// MultiHeadAttention_23605140258931
// MI455X (gfx1250) — compile-verified, matrix operand order checked by reading, not run on hardware
//
#include <hip/hip_runtime.h>
#include <hip/hip_bf16.h>

typedef __attribute__((ext_vector_type(16))) __bf16 bf16x16;
typedef __attribute__((ext_vector_type(4)))  __bf16 bf16x4;
typedef __attribute__((ext_vector_type(8)))  float  v8f;

#define C_DIM 1024
#define H_DIM 16
#define K_DIM 64
#define T_DIM 1024
#define B_DIM 4
#define WIN   4
#define NREL  9

__device__ __forceinline__ v8f v8f_zero() {
    v8f z;
#pragma unroll
    for (int e = 0; e < 8; ++e) z[e] = 0.0f;
    return z;
}

// Load a 16x32 bf16 A/B fragment slice from an LDS row.
// ISA layout (16-bit A 16x32): lane l: m = l%16, half = l/16,
// elems 0..7  -> k = k0 + half*8 + e ; elems 8..15 -> k = k0 + 16 + half*8 + (e-8)
__device__ __forceinline__ bf16x16 load_frag(const __bf16* row, int k0, int hf) {
    const __bf16* p = row + k0 + hf * 8;
    bf16x16 r;
#pragma unroll
    for (int e = 0; e < 8; ++e) { r[e] = p[e]; r[e + 8] = p[e + 16]; }
    return r;
}

__device__ __forceinline__ v8f wmma_bf16(bf16x16 a, bf16x16 b, v8f c) {
    return __builtin_amdgcn_wmma_f32_16x16x32_bf16(false, a, false, b, (short)0, c, false, false);
}

// CDNA5 async global->LDS copy (ASYNCcnt-tracked DMA, no VGPR round-trip).
// VFLAT encoding: VDST = LDS byte address VGPR, ADDR = 64-bit global address.
__device__ __forceinline__ void async_b128(unsigned lds_off, const void* g) {
    asm volatile("global_load_async_to_lds_b128 %0, %1, off"
                 :: "v"(lds_off), "v"((unsigned long long)(size_t)g)
                 : "memory");
}
#define WAIT_ASYNC(imm) asm volatile("s_wait_asynccnt " #imm ::: "memory")

// ---------------------------------------------------------------------------
// Projection GEMM: Out[m,n] = (sum_c W[m,c] * X[b,c,n] + bias[m]) * scale
// out_mode: 0 = bf16 [B,H,T,K], 1 = bf16 [B,C,T], 2 = f32 [B,C,T]
// Tile: 128x128x32, 256 threads = 8 waves (2x4), each wave 64x32 = 8 WMMA accs.
// ---------------------------------------------------------------------------
__global__ __launch_bounds__(256)
void proj_gemm_kernel(const float* __restrict__ Wmat, const float* __restrict__ X,
                      const float* __restrict__ bias, void* __restrict__ out,
                      float scale, int out_mode) {
    constexpr int BM = 128, BN = 128, BK = 32, LDA = BK + 8;
    __shared__ __align__(16) __bf16 Asm[BM * LDA];
    __shared__ __align__(16) __bf16 Bsm[BN * LDA];

    const int tid  = threadIdx.x;
    const int lane = tid & 31;
    const int wave = tid >> 5;
    const int hf   = lane >> 4;
    const int ln   = lane & 15;
    const int wm   = wave >> 2;     // 0..1
    const int wn   = wave & 3;      // 0..3
    const int m0   = blockIdx.x * BM;
    const int n0   = blockIdx.y * BN;
    const int b    = blockIdx.z;
    const float* Xb = X + (size_t)b * C_DIM * T_DIM;

    v8f acc[4][2];
#pragma unroll
    for (int i = 0; i < 4; ++i)
#pragma unroll
        for (int j = 0; j < 2; ++j) acc[i][j] = v8f_zero();

    for (int k0 = 0; k0 < C_DIM; k0 += BK) {
        __syncthreads();
        // Stage A: W[m0+r][k0+c4..c4+3] -> bf16x4 (float4 global loads)
        for (int i = tid; i < BM * BK / 4; i += 256) {
            int r = i >> 3, c4 = (i & 7) * 4;
            float4 w = *(const float4*)&Wmat[(size_t)(m0 + r) * C_DIM + k0 + c4];
            bf16x4 w4;
            w4[0] = (__bf16)w.x; w4[1] = (__bf16)w.y; w4[2] = (__bf16)w.z; w4[3] = (__bf16)w.w;
            *(bf16x4*)&Asm[r * LDA + c4] = w4;
        }
        // Stage B transposed: X[k0+c][n0+j..j+3] -> Bsm[j..j+3][c]
        for (int i = tid; i < BK * BN / 4; i += 256) {
            int c = i >> 5, j = (i & 31) * 4;
            float4 xv = *(const float4*)&Xb[(size_t)(k0 + c) * T_DIM + n0 + j];
            Bsm[(j + 0) * LDA + c] = (__bf16)xv.x;
            Bsm[(j + 1) * LDA + c] = (__bf16)xv.y;
            Bsm[(j + 2) * LDA + c] = (__bf16)xv.z;
            Bsm[(j + 3) * LDA + c] = (__bf16)xv.w;
        }
        __syncthreads();
#pragma unroll
        for (int mt = 0; mt < 4; ++mt) {
            bf16x16 af = load_frag(&Asm[(wm * 64 + mt * 16 + ln) * LDA], 0, hf);
#pragma unroll
            for (int nt = 0; nt < 2; ++nt) {
                bf16x16 bfr = load_frag(&Bsm[(wn * 32 + nt * 16 + ln) * LDA], 0, hf);
                acc[mt][nt] = wmma_bf16(af, bfr, acc[mt][nt]);
            }
        }
    }

    // Epilogue. C/D layout: row m = j + 8*hf, col n = ln.
#pragma unroll
    for (int mt = 0; mt < 4; ++mt) {
#pragma unroll
        for (int nt = 0; nt < 2; ++nt) {
#pragma unroll
            for (int j = 0; j < 8; ++j) {
                int m = m0 + wm * 64 + mt * 16 + j + 8 * hf;
                int n = n0 + wn * 32 + nt * 16 + ln;
                float v = (acc[mt][nt][j] + bias[m]) * scale;
                if (out_mode == 0) {
                    int h = m >> 6, k = m & 63;
                    ((__bf16*)out)[(((size_t)b * H_DIM + h) * T_DIM + n) * K_DIM + k] = (__bf16)v;
                } else if (out_mode == 1) {
                    ((__bf16*)out)[((size_t)b * C_DIM + m) * T_DIM + n] = (__bf16)v;
                } else {
                    ((float*)out)[((size_t)b * C_DIM + m) * T_DIM + n] = v;
                }
            }
        }
    }
}

// ---------------------------------------------------------------------------
// Fused flash attention + relative-position band terms.
// Block = 128 threads (4 waves); block handles (b, h, 64-row t-tile).
// Q,K bf16 [B,H,T,K]; V bf16 [B,C,T] (== V^T per head); out fp32 [B,C,T].
// K/V tiles stream in via async global->LDS DMA, double-buffered so the DMA
// of tile i+1 overlaps the WMMAs of tile i (per-wave 8 async ops per tile;
// s_wait_asynccnt 8 retires exactly the current tile since completion is
// in-order).
// ---------------------------------------------------------------------------
constexpr int LDQ = 72;   // padded LDS row stride (bf16), 144 B = 16B aligned

__global__ __launch_bounds__(128)
void flash_rel_attn_kernel(const __bf16* __restrict__ Q, const __bf16* __restrict__ Kg,
                           const __bf16* __restrict__ Vg, const float* __restrict__ relk,
                           const float* __restrict__ relv, float* __restrict__ Oattn) {
    __shared__ __align__(16) __bf16 Qs [64 * LDQ];
    __shared__ __align__(16) __bf16 Ks [2][64 * LDQ];
    __shared__ __align__(16) __bf16 Vts[2][64 * LDQ];
    __shared__ __align__(16) __bf16 Ps [64 * LDQ];
    __shared__ float relk_s[NREL * K_DIM];
    __shared__ float relv_s[NREL * K_DIM];
    __shared__ float rband [64 * NREL];
    __shared__ float pband [64 * NREL];
    __shared__ float mstat[64], lstat[64];

    const int tid  = threadIdx.x;
    const int lane = tid & 31;
    const int wave = tid >> 5;
    const int hf   = lane >> 4;
    const int ln   = lane & 15;
    const int t0   = blockIdx.x * 64;
    const int h    = blockIdx.y;
    const int b    = blockIdx.z;

    const __bf16* Qbase = Q  + (((size_t)b * H_DIM + h) * T_DIM + t0) * K_DIM;
    const __bf16* Kbh   = Kg + ((size_t)b * H_DIM + h) * (size_t)T_DIM * K_DIM;
    const __bf16* Vbh   = Vg + ((size_t)b * C_DIM + h * K_DIM) * (size_t)T_DIM;

    // Async-stage Q tile (4 ops/wave) then K/V tile 0 (8 ops/wave).
    for (int i = tid; i < 64 * 8; i += 128) {
        int r = i >> 3, cq = i & 7;
        async_b128((unsigned)(size_t)(Qs + r * LDQ) + cq * 16,
                   Qbase + (size_t)r * K_DIM + cq * 8);
    }
    for (int i = tid; i < 64 * 8; i += 128) {
        int r = i >> 3, cq = i & 7;
        async_b128((unsigned)(size_t)(Ks[0] + r * LDQ) + cq * 16,
                   Kbh + (size_t)r * K_DIM + cq * 8);
        async_b128((unsigned)(size_t)(Vts[0] + r * LDQ) + cq * 16,
                   Vbh + (size_t)r * T_DIM + cq * 8);
    }
    for (int i = tid; i < NREL * K_DIM; i += 128) { relk_s[i] = relk[i]; relv_s[i] = relv[i]; }
    WAIT_ASYNC(0x8);     // Q done; tile0's 8 ops may still be in flight
    __syncthreads();

    // rband[t][d] = (scaled q_t) . rel_k[d]   (Q already carries 1/sqrt(K))
    for (int i = tid; i < 64 * NREL; i += 128) {
        int tl = i / NREL, d = i % NREL;
        float s = 0.0f;
        for (int k = 0; k < K_DIM; ++k) s += (float)Qs[tl * LDQ + k] * relk_s[d * K_DIM + k];
        rband[i] = s;
    }

    v8f Sacc[4], Oacc[4];
    float mrow[8], lrow[8];
#pragma unroll
    for (int kt = 0; kt < 4; ++kt) Oacc[kt] = v8f_zero();
#pragma unroll
    for (int j = 0; j < 8; ++j) { mrow[j] = -3.0e38f; lrow[j] = 0.0f; }

    for (int it = 0; it < T_DIM / 64; ++it) {
        const int buf = it & 1;
        const int s0  = it * 64;
        __syncthreads();   // all waves done reading buf^1 (tile it-1)
        if (it + 1 < T_DIM / 64) {
            const int s1 = s0 + 64;
            for (int i = tid; i < 64 * 8; i += 128) {
                int r = i >> 3, cq = i & 7;
                async_b128((unsigned)(size_t)(Ks[buf ^ 1] + r * LDQ) + cq * 16,
                           Kbh + (size_t)(s1 + r) * K_DIM + cq * 8);
                async_b128((unsigned)(size_t)(Vts[buf ^ 1] + r * LDQ) + cq * 16,
                           Vbh + (size_t)r * T_DIM + s1 + cq * 8);
            }
            WAIT_ASYNC(0x8);   // retire tile it (oldest 8); tile it+1 stays in flight
        } else {
            WAIT_ASYNC(0x0);
        }
        __syncthreads();       // tile it fully visible in LDS

        // S = Q K^T  (16 t-rows per wave x 64 s-cols, K-depth 64 = 2 WMMA steps)
#pragma unroll
        for (int st = 0; st < 4; ++st) {
            v8f s8 = v8f_zero();
#pragma unroll
            for (int kk = 0; kk < K_DIM; kk += 32) {
                bf16x16 af  = load_frag(&Qs[(wave * 16 + ln) * LDQ], kk, hf);
                bf16x16 bfr = load_frag(&Ks[buf][(st * 16 + ln) * LDQ], kk, hf);
                s8 = wmma_bf16(af, bfr, s8);
            }
            Sacc[st] = s8;
        }

        // Band add + online softmax + P to LDS (bf16) — per accumulator row j.
#pragma unroll
        for (int j = 0; j < 8; ++j) {
            const int tl = wave * 16 + j + 8 * hf;
            const int t  = t0 + tl;
            float rmax = -3.0e38f;
#pragma unroll
            for (int st = 0; st < 4; ++st) {
                int s = s0 + st * 16 + ln;
                int d = s - t + WIN;
                float v = Sacc[st][j];
                if (d >= 0 && d < NREL) v += rband[tl * NREL + d];
                Sacc[st][j] = v;
                rmax = fmaxf(rmax, v);
            }
            rmax = fmaxf(rmax, __shfl_xor(rmax, 1, 32));
            rmax = fmaxf(rmax, __shfl_xor(rmax, 2, 32));
            rmax = fmaxf(rmax, __shfl_xor(rmax, 4, 32));
            rmax = fmaxf(rmax, __shfl_xor(rmax, 8, 32));
            float mnew  = fmaxf(mrow[j], rmax);
            float alpha = __expf(mrow[j] - mnew);
            float rsum  = 0.0f;
#pragma unroll
            for (int st = 0; st < 4; ++st) {
                float p = __expf(Sacc[st][j] - mnew);
                Sacc[st][j] = p;
                rsum += p;
            }
            rsum += __shfl_xor(rsum, 1, 32);
            rsum += __shfl_xor(rsum, 2, 32);
            rsum += __shfl_xor(rsum, 4, 32);
            rsum += __shfl_xor(rsum, 8, 32);
            lrow[j] = lrow[j] * alpha + rsum;
            mrow[j] = mnew;
#pragma unroll
            for (int kt = 0; kt < 4; ++kt) Oacc[kt][j] *= alpha;
#pragma unroll
            for (int st = 0; st < 4; ++st)
                Ps[tl * LDQ + st * 16 + ln] = (__bf16)Sacc[st][j];
        }

        // O += P V   (contraction over s, V^T staged so B-frags are row slices)
#pragma unroll
        for (int kt = 0; kt < 4; ++kt) {
#pragma unroll
            for (int ss = 0; ss < 64; ss += 32) {
                bf16x16 af  = load_frag(&Ps[(wave * 16 + ln) * LDQ], ss, hf);
                bf16x16 bfr = load_frag(&Vts[buf][(kt * 16 + ln) * LDQ], ss, hf);
                Oacc[kt] = wmma_bf16(af, bfr, Oacc[kt]);
            }
        }
    }

    // Publish per-row stats for the band epilogue.
    if (ln == 0) {
#pragma unroll
        for (int j = 0; j < 8; ++j) {
            int tl = wave * 16 + j + 8 * hf;
            mstat[tl] = mrow[j];
            lstat[tl] = lrow[j];
        }
    }
    __syncthreads();

    // Recompute the 9 band scores per row -> normalized band probabilities.
    for (int i = tid; i < 64 * NREL; i += 128) {
        int tl = i / NREL, d = i % NREL;
        int t = t0 + tl;
        int s = t + d - WIN;
        float p = 0.0f;
        if (s >= 0 && s < T_DIM) {
            float sc = 0.0f;
            const __bf16* kr = Kbh + (size_t)s * K_DIM;
            for (int k = 0; k < K_DIM; ++k) sc += (float)Qs[tl * LDQ + k] * (float)kr[k];
            sc += rband[i];
            p = __expf(sc - mstat[tl]) / lstat[tl];
        }
        pband[i] = p;
    }
    __syncthreads();

    // out[t][kv] = O/l + sum_d pband[t][d] * rel_v[d][kv];  store [B,C,T] fp32.
#pragma unroll
    for (int kt = 0; kt < 4; ++kt) {
        const int kv = kt * 16 + ln;
#pragma unroll
        for (int j = 0; j < 8; ++j) {
            int tl = wave * 16 + j + 8 * hf;
            float v = Oacc[kt][j] / lrow[j];
#pragma unroll
            for (int d = 0; d < NREL; ++d)
                v += pband[tl * NREL + d] * relv_s[d * K_DIM + kv];
            Oattn[((size_t)b * C_DIM + h * K_DIM + kv) * T_DIM + (t0 + tl)] = v;
        }
    }
}

// ---------------------------------------------------------------------------
extern "C" void kernel_launch(void* const* d_in, const int* in_sizes, int n_in,
                              void* d_out, int out_size, void* d_ws, size_t ws_size,
                              hipStream_t stream) {
    const float* x    = (const float*)d_in[0];
    const float* c    = (const float*)d_in[1];
    const float* Wq   = (const float*)d_in[2];
    const float* bq   = (const float*)d_in[3];
    const float* Wk   = (const float*)d_in[4];
    const float* bk   = (const float*)d_in[5];
    const float* Wv   = (const float*)d_in[6];
    const float* bv   = (const float*)d_in[7];
    const float* Wo   = (const float*)d_in[8];
    const float* bo   = (const float*)d_in[9];
    const float* relk = (const float*)d_in[10];
    const float* relv = (const float*)d_in[11];

    char* ws = (char*)d_ws;
    __bf16* Qbf = (__bf16*)(ws);                               // 8 MB  [B,H,T,K]
    __bf16* Kbf = (__bf16*)(ws + ((size_t)8  << 20));          // 8 MB  [B,H,T,K]
    __bf16* Vbf = (__bf16*)(ws + ((size_t)16 << 20));          // 8 MB  [B,C,T]
    float*  Oat = (float*) (ws + ((size_t)24 << 20));          // 16 MB [B,C,T]

    const float qscale = 0.125f;   // 1/sqrt(K_DIM)
    dim3 gg(C_DIM / 128, T_DIM / 128, B_DIM), bb(256);
    proj_gemm_kernel<<<gg, bb, 0, stream>>>(Wq, x, bq, (void*)Qbf, qscale, 0);
    proj_gemm_kernel<<<gg, bb, 0, stream>>>(Wk, c, bk, (void*)Kbf, 1.0f,   0);
    proj_gemm_kernel<<<gg, bb, 0, stream>>>(Wv, c, bv, (void*)Vbf, 1.0f,   1);

    dim3 gf(T_DIM / 64, H_DIM, B_DIM);
    flash_rel_attn_kernel<<<gf, dim3(128), 0, stream>>>(Qbf, Kbf, Vbf, relk, relv, Oat);

    proj_gemm_kernel<<<gg, bb, 0, stream>>>(Wo, Oat, bo, d_out, 1.0f, 2);
}
